// QGNN_4758823764712
// MI455X (gfx1250) — compile-verified, matrix operand order checked by reading, not run on hardware
//
#include <hip/hip_runtime.h>
#include <hip/hip_bf16.h>

typedef __attribute__((ext_vector_type(16))) _Float16 v16h;
typedef __attribute__((ext_vector_type(8)))  _Float16 v8h;
typedef __attribute__((ext_vector_type(8)))  float    v8f;

#define WMMA16(a, b, c) __builtin_amdgcn_wmma_f32_16x16x32_f16( \
    false, (a), false, (b), (short)0, (c), false, false)

// ---------------------------------------------------------------------------
// atomic float max via int/uint monotonic mapping (valid with -inf init)
// ---------------------------------------------------------------------------
__device__ inline void atomicMaxFloat(float* addr, float v) {
    if (v >= 0.0f) atomicMax((int*)addr, __float_as_int(v));
    else           atomicMin((unsigned int*)addr, __float_as_uint(v));
}

// ---------------------------------------------------------------------------
// Swizzle a row-major fp32 [K x Ncols] weight matrix into f16 B-fragments:
// frag f = ((kt*njt + jt)*32 + lane)*16 + i
//   k = kt*32 + (lane>=16 ? 8 : 0) + (i%8) + (i/8)*16 ;  c = jt*16 + lane%16
// Each lane's 16 halfs land contiguous (32B) -> 2x global_load_b128 at use.
// ---------------------------------------------------------------------------
__global__ void swizzleB_kernel(const float* __restrict__ src,
                                _Float16* __restrict__ dst, int K, int Ncols) {
    int f = blockIdx.x * blockDim.x + threadIdx.x;
    if (f >= K * Ncols) return;
    int njt  = Ncols >> 4;
    int i    = f & 15;
    int lane = (f >> 4) & 31;
    int jt   = (f >> 9) % njt;
    int kt   = (f >> 9) / njt;
    int k = kt * 32 + ((lane >> 4) << 3) + (i & 7) + ((i >> 3) << 4);
    int c = jt * 16 + (lane & 15);
    dst[f] = (_Float16)src[(size_t)k * Ncols + c];
}

// Fold flag row of msg_W1 (row 256) + msg_b1 into per-direction bias vectors.
__global__ void prep_bias_kernel(const float* __restrict__ msg_W1,
                                 const float* __restrict__ msg_b1,
                                 float* __restrict__ bias_up,
                                 float* __restrict__ bias_down) {
    int i = blockIdx.x * blockDim.x + threadIdx.x;
    if (i < 256) {
        bias_up[i]   = msg_b1[i] + msg_W1[256 * 256 + i];  // flag = 1.0
        bias_down[i] = msg_b1[i];                          // flag = 0.0
    }
}

__global__ void init_seg_kernel(float* __restrict__ segmax,
                                float* __restrict__ segsum, int n) {
    int i = blockIdx.x * blockDim.x + threadIdx.x;
    if (i < n) { segmax[i] = -3.0e38f; segsum[i] = 0.0f; }
}

// Zero the message columns [128..384) of the output.
__global__ void zero_out_kernel(float* __restrict__ out, int n) {
    int i = blockIdx.x * blockDim.x + threadIdx.x;
    if (i < n * 256) {
        int r = i >> 8, c = i & 255;
        out[(size_t)r * 384 + 128 + c] = 0.0f;
    }
}

// ---------------------------------------------------------------------------
// Node MLP: h = relu(x @ W1 + b1) @ W2 + b2.  One wave per 16-row tile.
// ---------------------------------------------------------------------------
__global__ __launch_bounds__(128) void node_mlp_kernel(
    const float* __restrict__ x, const float* __restrict__ b1,
    const float* __restrict__ b2, const _Float16* __restrict__ W1sw,
    const _Float16* __restrict__ W2sw, float* __restrict__ out,
    _Float16* __restrict__ h16, int n) {
    __shared__ _Float16 lds_m1[4][16 * 128];  // 16KB
    int wave = threadIdx.x >> 5, lane = threadIdx.x & 31;
    int t = blockIdx.x * 4 + wave;
    int ntiles = n >> 4;
    if (t >= ntiles) return;  // whole-wave uniform
    int m0 = t * 16;
    int row = lane & 15, hi = lane >> 4, kb = hi << 3, col = lane & 15;

    // A fragments from x (fp32 -> f16), K = 128 -> 4 fragments
    v16h afr[4];
    const float* xr = x + (size_t)(m0 + row) * 128;
#pragma unroll
    for (int kf = 0; kf < 4; ++kf) {
        int k0 = kf * 32 + kb;
#pragma unroll
        for (int i = 0; i < 8; ++i) {
            afr[kf][i]     = (_Float16)xr[k0 + i];
            afr[kf][8 + i] = (_Float16)xr[k0 + 16 + i];
        }
    }
    _Float16* m1 = lds_m1[wave];
    const v16h* W1f = (const v16h*)W1sw;  // nkt=4, njt=8
#pragma unroll
    for (int j = 0; j < 8; ++j) {
        float bv = b1[j * 16 + col];
        v8f acc;
#pragma unroll
        for (int i = 0; i < 8; ++i) acc[i] = bv;
#pragma unroll
        for (int kf = 0; kf < 4; ++kf)
            acc = WMMA16(afr[kf], W1f[(kf * 8 + j) * 32 + lane], acc);
#pragma unroll
        for (int r = 0; r < 8; ++r) {
            float v = acc[r];
            m1[(r + hi * 8) * 128 + j * 16 + col] =
                (_Float16)(v > 0.0f ? v : 0.0f);
        }
    }
    // re-fragment layer1 output from LDS (same-wave DS ordering)
    v16h a2[4];
    const v8h* m1v = (const v8h*)(m1 + row * 128);
#pragma unroll
    for (int kf = 0; kf < 4; ++kf) {
        int k0 = kf * 32 + kb;
        v8h lo = m1v[k0 >> 3], hp = m1v[(k0 + 16) >> 3];
#pragma unroll
        for (int i = 0; i < 8; ++i) { a2[kf][i] = lo[i]; a2[kf][8 + i] = hp[i]; }
    }
    const v16h* W2f = (const v16h*)W2sw;
#pragma unroll
    for (int j = 0; j < 8; ++j) {
        float bv = b2[j * 16 + col];
        v8f acc;
#pragma unroll
        for (int i = 0; i < 8; ++i) acc[i] = bv;
#pragma unroll
        for (int kf = 0; kf < 4; ++kf)
            acc = WMMA16(a2[kf], W2f[(kf * 8 + j) * 32 + lane], acc);
#pragma unroll
        for (int r = 0; r < 8; ++r) {
            int rg = m0 + r + hi * 8;
            float v = acc[r];
            out[(size_t)rg * 384 + j * 16 + col] = v;
            h16[(size_t)rg * 128 + j * 16 + col] = (_Float16)v;
        }
    }
}

// ---------------------------------------------------------------------------
// Edge pass 1: per-edge message MLP + attention score.  One wave / 16 edges.
// ---------------------------------------------------------------------------
__global__ __launch_bounds__(128) void edge_pass1_kernel(
    const _Float16* __restrict__ h16, const int* __restrict__ src,
    const int* __restrict__ dst, const _Float16* __restrict__ mW1sw,
    const float* __restrict__ biasvec, const _Float16* __restrict__ mW2sw,
    const float* __restrict__ msg_b2, const _Float16* __restrict__ aW1sw,
    const float* __restrict__ a_b1, const float* __restrict__ aW2,
    const float* __restrict__ a_b2, _Float16* __restrict__ msgs16,
    float* __restrict__ scores, float* __restrict__ segmax, int ne) {
    __shared__ _Float16 lds_buf[4][16 * 256];  // 32KB (8KB/wave, reused)
    int wave = threadIdx.x >> 5, lane = threadIdx.x & 31;
    int t = blockIdx.x * 4 + wave;
    int ntiles = ne >> 4;
    if (t >= ntiles) return;
    int e0 = t * 16;
    int row = lane & 15, hi = lane >> 4, kb = hi << 3, col = lane & 15;

    int d_idx = dst[e0 + row];
    int s_idx = src[e0 + row];
    const _Float16* hd = h16 + (size_t)d_idx * 128;
    const _Float16* hs = h16 + (size_t)s_idx * 128;

    // gather A fragments: K=256 (cols 0..127 = h[dst], 128..255 = h[src])
    v16h afr[8];
#pragma unroll
    for (int kf = 0; kf < 8; ++kf) {
        const _Float16* base = (kf < 4) ? hd : hs;
        int kc = (kf & 3) * 32 + kb;
        v8h lo = *(const v8h*)(base + kc);
        v8h hp = *(const v8h*)(base + kc + 16);
#pragma unroll
        for (int i = 0; i < 8; ++i) { afr[kf][i] = lo[i]; afr[kf][8 + i] = hp[i]; }
    }

    // GEMM1: [16x256] @ [256x256] -> relu -> LDS f16
    _Float16* m1 = lds_buf[wave];
    const v16h* W1f = (const v16h*)mW1sw;  // nkt=8, njt=16
#pragma unroll
    for (int j = 0; j < 16; ++j) {
        float bv = biasvec[j * 16 + col];
        v8f acc;
#pragma unroll
        for (int i = 0; i < 8; ++i) acc[i] = bv;
#pragma unroll
        for (int kf = 0; kf < 8; ++kf)
            acc = WMMA16(afr[kf], W1f[(kf * 16 + j) * 32 + lane], acc);
#pragma unroll
        for (int r = 0; r < 8; ++r) {
            float v = acc[r];
            m1[(r + hi * 8) * 256 + j * 16 + col] =
                (_Float16)(v > 0.0f ? v : 0.0f);
        }
    }
    // preload GEMM2 A fragments, then LDS buffer is reused
    v16h a2[8];
    {
        const v8h* m1v = (const v8h*)(m1 + row * 256);
#pragma unroll
        for (int kf = 0; kf < 8; ++kf) {
            int k0 = kf * 32 + kb;
            v8h lo = m1v[k0 >> 3], hp = m1v[(k0 + 16) >> 3];
#pragma unroll
            for (int i = 0; i < 8; ++i) { a2[kf][i] = lo[i]; a2[kf][8 + i] = hp[i]; }
        }
    }
    _Float16* m2 = m1;                       // 16x128 f16 (4KB)
    float* s1 = (float*)(m1 + 16 * 128);     // 16x64 f32  (4KB)

    // GEMM2: [16x256] @ [256x128] -> msgs (global f16 + LDS f16)
    const v16h* W2f = (const v16h*)mW2sw;  // nkt=8, njt=8
#pragma unroll
    for (int j = 0; j < 8; ++j) {
        float bv = msg_b2[j * 16 + col];
        v8f acc;
#pragma unroll
        for (int i = 0; i < 8; ++i) acc[i] = bv;
#pragma unroll
        for (int kf = 0; kf < 8; ++kf)
            acc = WMMA16(a2[kf], W2f[(kf * 8 + j) * 32 + lane], acc);
#pragma unroll
        for (int r = 0; r < 8; ++r) {
            int er = e0 + r + hi * 8;
            _Float16 v = (_Float16)acc[r];
            msgs16[(size_t)er * 128 + j * 16 + col] = v;
            m2[(r + hi * 8) * 128 + j * 16 + col] = v;
        }
    }
    // score GEMM: [16x128] @ [128x64] -> tanh -> LDS f32
    v16h a3[4];
    {
        const v8h* m2v = (const v8h*)(m2 + row * 128);
#pragma unroll
        for (int kf = 0; kf < 4; ++kf) {
            int k0 = kf * 32 + kb;
            v8h lo = m2v[k0 >> 3], hp = m2v[(k0 + 16) >> 3];
#pragma unroll
            for (int i = 0; i < 8; ++i) { a3[kf][i] = lo[i]; a3[kf][8 + i] = hp[i]; }
        }
    }
    const v16h* A1f = (const v16h*)aW1sw;  // nkt=4, njt=4
#pragma unroll
    for (int j = 0; j < 4; ++j) {
        float bv = a_b1[j * 16 + col];
        v8f acc;
#pragma unroll
        for (int i = 0; i < 8; ++i) acc[i] = bv;
#pragma unroll
        for (int kf = 0; kf < 4; ++kf)
            acc = WMMA16(a3[kf], A1f[(kf * 4 + j) * 32 + lane], acc);
#pragma unroll
        for (int r = 0; r < 8; ++r)
            s1[(r + hi * 8) * 64 + j * 16 + col] = tanhf(acc[r]);
    }
    // final score: 64-wide dot with aW2, one lane per edge
    if (lane < 16) {
        float sc = a_b2[0];
#pragma unroll 8
        for (int a = 0; a < 64; ++a) sc += s1[row * 64 + a] * aW2[a];
        scores[e0 + row] = sc;
        atomicMaxFloat(&segmax[d_idx], sc);
    }
}

__global__ void edge_pass2_kernel(const float* __restrict__ scores,
                                  const int* __restrict__ dst,
                                  const float* __restrict__ segmax,
                                  float* __restrict__ expw,
                                  float* __restrict__ segsum, int ne) {
    int e = blockIdx.x * blockDim.x + threadIdx.x;
    if (e >= ne) return;
    int d = dst[e];
    float m = segmax[d];
    if (!(m > -1.0e37f)) m = 0.0f;
    float ex = __expf(scores[e] - m);
    expw[e] = ex;
    atomicAdd(&segsum[d], ex);
}

__global__ void edge_pass3_kernel(const _Float16* __restrict__ msgs16,
                                  const float* __restrict__ expw,
                                  const float* __restrict__ segsum,
                                  const int* __restrict__ dst,
                                  float* __restrict__ out, int colbase, int ne) {
    int idx = blockIdx.x * blockDim.x + threadIdx.x;  // ne * 4 work items
    if (idx >= ne * 4) return;
    int e = idx >> 2, c0 = (idx & 3) * 32;
    int d = dst[e];
    float s = segsum[d];
    float w = expw[e] / (s > 0.0f ? s : 1.0f);
    const _Float16* mrow = msgs16 + (size_t)e * 128 + c0;
    float* orow = out + (size_t)d * 384 + colbase + c0;
#pragma unroll
    for (int i = 0; i < 32; ++i) atomicAdd(&orow[i], w * (float)mrow[i]);
}

// ---------------------------------------------------------------------------
extern "C" void kernel_launch(void* const* d_in, const int* in_sizes, int n_in,
                              void* d_out, int out_size, void* d_ws,
                              size_t ws_size, hipStream_t stream) {
    (void)n_in; (void)out_size; (void)ws_size;
    const float* x_nodes = (const float*)d_in[0];
    const float* node_W1 = (const float*)d_in[1];
    const float* node_b1 = (const float*)d_in[2];
    const float* node_W2 = (const float*)d_in[3];
    const float* node_b2 = (const float*)d_in[4];
    const float* msg_W1  = (const float*)d_in[5];
    const float* msg_b1  = (const float*)d_in[6];
    const float* msg_W2  = (const float*)d_in[7];
    const float* msg_b2  = (const float*)d_in[8];
    const float* ap_W1   = (const float*)d_in[9];
    const float* ap_b1   = (const float*)d_in[10];
    const float* ap_W2   = (const float*)d_in[11];
    const float* ap_b2   = (const float*)d_in[12];
    const float* ac_W1   = (const float*)d_in[13];
    const float* ac_b1   = (const float*)d_in[14];
    const float* ac_W2   = (const float*)d_in[15];
    const float* ac_b2   = (const float*)d_in[16];
    const int* p_src = (const int*)d_in[17];
    const int* p_dst = (const int*)d_in[18];
    const int* c_src = (const int*)d_in[19];
    const int* c_dst = (const int*)d_in[20];
    float* out = (float*)d_out;

    const int N_ = in_sizes[0] / 128;  // 50000
    const int E_ = in_sizes[17];       // 200000

    // workspace layout (256B aligned)
    size_t off = 0;
    auto alloc = [&](size_t bytes) {
        size_t o = off; off = (off + bytes + 255) & ~size_t(255); return o;
    };
    char* w = (char*)d_ws;
    _Float16* h16    = (_Float16*)(w + alloc((size_t)N_ * 128 * 2));
    _Float16* nW1sw  = (_Float16*)(w + alloc(128 * 128 * 2));
    _Float16* nW2sw  = (_Float16*)(w + alloc(128 * 128 * 2));
    _Float16* mW1sw  = (_Float16*)(w + alloc(256 * 256 * 2));
    _Float16* mW2sw  = (_Float16*)(w + alloc(256 * 128 * 2));
    _Float16* apW1sw = (_Float16*)(w + alloc(128 * 64 * 2));
    _Float16* acW1sw = (_Float16*)(w + alloc(128 * 64 * 2));
    float* bias_up   = (float*)(w + alloc(256 * 4));
    float* bias_down = (float*)(w + alloc(256 * 4));
    _Float16* msgs16 = (_Float16*)(w + alloc((size_t)E_ * 128 * 2));
    float* scores    = (float*)(w + alloc((size_t)E_ * 4));
    float* expw      = (float*)(w + alloc((size_t)E_ * 4));
    float* segmax    = (float*)(w + alloc((size_t)N_ * 4));
    float* segsum    = (float*)(w + alloc((size_t)N_ * 4));

    // --- weight prep ---
    swizzleB_kernel<<<(128 * 128 + 255) / 256, 256, 0, stream>>>(node_W1, nW1sw, 128, 128);
    swizzleB_kernel<<<(128 * 128 + 255) / 256, 256, 0, stream>>>(node_W2, nW2sw, 128, 128);
    swizzleB_kernel<<<(256 * 256 + 255) / 256, 256, 0, stream>>>(msg_W1, mW1sw, 256, 256);
    swizzleB_kernel<<<(256 * 128 + 255) / 256, 256, 0, stream>>>(msg_W2, mW2sw, 256, 128);
    swizzleB_kernel<<<(128 * 64 + 255) / 256, 256, 0, stream>>>(ap_W1, apW1sw, 128, 64);
    swizzleB_kernel<<<(128 * 64 + 255) / 256, 256, 0, stream>>>(ac_W1, acW1sw, 128, 64);
    prep_bias_kernel<<<1, 256, 0, stream>>>(msg_W1, msg_b1, bias_up, bias_down);
    zero_out_kernel<<<(N_ * 256 + 255) / 256, 256, 0, stream>>>(out, N_);

    // --- node MLP ---
    int ntiles_n = N_ / 16;
    node_mlp_kernel<<<(ntiles_n + 3) / 4, 128, 0, stream>>>(
        x_nodes, node_b1, node_b2, nW1sw, nW2sw, out, h16, N_);

    int ntiles_e = E_ / 16;
    // --- direction up (parent edges, flag=1) -> cols [128,256) ---
    init_seg_kernel<<<(N_ + 255) / 256, 256, 0, stream>>>(segmax, segsum, N_);
    edge_pass1_kernel<<<(ntiles_e + 3) / 4, 128, 0, stream>>>(
        h16, p_src, p_dst, mW1sw, bias_up, mW2sw, msg_b2, apW1sw, ap_b1, ap_W2,
        ap_b2, msgs16, scores, segmax, E_);
    edge_pass2_kernel<<<(E_ + 255) / 256, 256, 0, stream>>>(
        scores, p_dst, segmax, expw, segsum, E_);
    edge_pass3_kernel<<<(E_ * 4 + 255) / 256, 256, 0, stream>>>(
        msgs16, expw, segsum, p_dst, out, 128, E_);

    // --- direction down (child edges, flag=0) -> cols [256,384) ---
    init_seg_kernel<<<(N_ + 255) / 256, 256, 0, stream>>>(segmax, segsum, N_);
    edge_pass1_kernel<<<(ntiles_e + 3) / 4, 128, 0, stream>>>(
        h16, c_src, c_dst, mW1sw, bias_down, mW2sw, msg_b2, acW1sw, ac_b1,
        ac_W2, ac_b2, msgs16, scores, segmax, E_);
    edge_pass2_kernel<<<(E_ + 255) / 256, 256, 0, stream>>>(
        scores, c_dst, segmax, expw, segsum, E_);
    edge_pass3_kernel<<<(E_ * 4 + 255) / 256, 256, 0, stream>>>(
        msgs16, expw, segsum, c_dst, out, 256, E_);
}
